// GConv_45243185496758
// MI455X (gfx1250) — matrix operand order repaired and checked by reading, not compiled, not run on hardware
//
#include <hip/hip_runtime.h>

// out[b,n,g,f] = sum_{h,k} f[b,n,h,k] * psi[(G-g+h)%G, k, f]
// Per output group g: GEMM out_g[M,F] = F[M,G*K] x Wg[G*K,F], Wg = block-rotated psi.
// Roofline: 68.7 GFLOP vs ~270 MB HBM (~12us floor at 23.3TB/s) -> bf16 WMMA
// 16x16x32 (~4.2M matrix issues) instead of f32 WMMA K=4 (~34M issues).
//
// Block = 8 wave32s; wave w owns output group g=w (circular psi permutation is
// just a per-wave LDS base). Whole psi lives in LDS as bf16 [g][f][k] (256KB,
// XOR-swizzled, B frag = 2x ds_load_b128/lane); f streamed in 32-row bf16
// tiles (64KB): two M-subtiles per wave so each B fragment feeds 2 WMMAs
// (1.25 LDS b128 loads per WMMA). LDS = exactly 320KB/WGP.

#define DEV __device__ __forceinline__

typedef __attribute__((ext_vector_type(16))) __bf16 v16bf;
typedef __attribute__((ext_vector_type(8)))  float  v8f;

constexpr int BB = 8, NN = 4096, G = 8, K = 128, F = 128;
constexpr int M   = BB * NN;           // 32768 output rows
constexpr int KH  = G * K;             // 1024 contraction length
constexpr int MT  = 32;                // rows per block tile (2 WMMA M-slices/wave)
constexpr int TPB = 4;                 // tiles per block (amortize psi staging)
constexpr int NBLK = (M / MT) / TPB;   // 256 blocks

DEV unsigned int bf16_rne(float a) {
  unsigned u = __float_as_uint(a);
  return (u + 0x7FFFu + ((u >> 16) & 1u)) >> 16;
}
DEV unsigned int pack2(float a, float b) {
  unsigned ua = __float_as_uint(a);
  unsigned ub = __float_as_uint(b);
  ua += 0x7FFFu + ((ua >> 16) & 1u);
  ub += 0x7FFFu + ((ub >> 16) & 1u);
  return (ub & 0xFFFF0000u) | (ua >> 16);
}

// Two independent ds_load_b128 -> one 16-element bf16 WMMA fragment.
DEV v16bf ld_frag2(const unsigned short* p0, const unsigned short* p1) {
  union { uint4 q[2]; v16bf v; } u;
  u.q[0] = *(const uint4*)p0;
  u.q[1] = *(const uint4*)p1;
  return u.v;
}

__global__ __launch_bounds__(256) void gconv_wmma(const float* __restrict__ fin,
                                                  const float* __restrict__ psi,
                                                  float* __restrict__ out) {
  // Exactly 320KB: swizzled, no pads. 16B-chunk XOR swizzle by row low nibble.
  __shared__ unsigned short s_psi[G * F * K];   // 256KB, [row=g*128+f][k] bf16
  __shared__ unsigned short s_f[MT * KH];       // 64KB,  [row][h*K+k] bf16

  const int tid  = threadIdx.x;
  const int lane = tid & 31;
  const int wv   = tid >> 5;        // wave id == output group g
  const int l15  = lane & 15;
  const int half = lane >> 4;

  // ---- stage whole psi once: global [h][k][f] f32 -> LDS [h][f][k] bf16 ----
  {
    const float4* p4 = (const float4*)psi;
    for (int i = tid; i < (G * K * F) / 4; i += 256) {
      float4 v = p4[i];
      int flat = i << 2;                   // [h][k][f] flat element index
      int f0 = flat & (F - 1);
      int kk = (flat >> 7) & (K - 1);
      int hh = flat >> 14;
      int c  = kk >> 3;                    // logical 16B chunk in row (0..15)
      int ko = kk & 7;
      unsigned short b0 = (unsigned short)bf16_rne(v.x);
      unsigned short b1 = (unsigned short)bf16_rne(v.y);
      unsigned short b2 = (unsigned short)bf16_rne(v.z);
      unsigned short b3 = (unsigned short)bf16_rne(v.w);
      s_psi[(hh * F + f0 + 0) * K + (((c ^ (f0 + 0)) & 15) << 3) + ko] = b0;
      s_psi[(hh * F + f0 + 1) * K + (((c ^ (f0 + 1)) & 15) << 3) + ko] = b1;
      s_psi[(hh * F + f0 + 2) * K + (((c ^ (f0 + 2)) & 15) << 3) + ko] = b2;
      s_psi[(hh * F + f0 + 3) * K + (((c ^ (f0 + 3)) & 15) << 3) + ko] = b3;
    }
  }

  for (int t = 0; t < TPB; ++t) {
    const int tile = blockIdx.x * TPB + t;
    const int m0   = tile * MT;

    __syncthreads();  // prior tile consumed; also orders psi staging at t==0

    // ---- stage f tile: 32 rows x 1024 f32 -> swizzled bf16 LDS ----
    {
      const float4* f4 = (const float4*)fin;
      const int base = m0 * (KH / 4);
      for (int i = tid; i < MT * (KH / 4); i += 256) {
        float4 v = f4[base + i];
        int row = i >> 8;                  // 256 float4 per row
        int hc  = i & 255;                 // 8-byte half-chunk index in row
        int c   = hc >> 1;                 // logical 16B chunk (0..127)
        int pc  = (c & 112) | ((c ^ row) & 15);
        uint2 w;
        w.x = pack2(v.x, v.y);
        w.y = pack2(v.z, v.w);
        *(uint2*)&s_f[row * KH + pc * 8 + (hc & 1) * 4] = w;  // ds_store_b64
      }
      if (t + 1 < TPB) {                   // global_prefetch_b8 the next tile
        const char* np = (const char*)(f4 + base + MT * (KH / 4));
        for (int i = tid; i < MT * (KH / 4); i += 256 * 8)
          __builtin_prefetch(np + (size_t)i * 16, 0, 1);
      }
    }
    __syncthreads();

    // ---- compute: wave wv owns g=wv; 2 M-subtiles x 8 F-tiles of f32 accum ----
    v8f acc0[8], acc1[8];
    const v8f vzero = {0.f, 0.f, 0.f, 0.f, 0.f, 0.f, 0.f, 0.f};
#pragma unroll
    for (int ft = 0; ft < 8; ++ft) { acc0[ft] = vzero; acc1[ft] = vzero; }

    // A layout: lanes 0-15 hold K {0..7,16..23}, lanes 16-31 {8..15,24..31};
    // both subtiles use row%16 == l15, so the swizzle nibble is l15 everywhere.
    const unsigned short* fr0 = &s_f[l15 * KH];
    const unsigned short* fr1 = &s_f[(l15 + 16) * KH];
    for (int h = 0; h < G; ++h) {
      const int hp = (G - wv + h) & (G - 1);   // circular weight-block permutation
      const unsigned short* pbase = &s_psi[(hp * F + l15) * K];
#pragma unroll
      for (int ks = 0; ks < 4; ++ks) {
        const int cA  = h * 16 + ks * 4 + half;        // logical chunks cA, cA+2
        const int oA0 = (((cA & 112) | ((cA ^ l15) & 15)) << 3);
        const int oA1 = (((cA & 112) | (((cA + 2) ^ l15) & 15)) << 3);
        v16bf A0 = ld_frag2(fr0 + oA0, fr0 + oA1);
        v16bf A1 = ld_frag2(fr1 + oA0, fr1 + oA1);
        const int cB  = ks * 4 + half;                 // logical chunks cB, cB+2
        const int oB0 = (((cB ^ l15) & 15) << 3);
        const int oB1 = ((((cB + 2) ^ l15) & 15) << 3);
#pragma unroll
        for (int ft = 0; ft < 8; ++ft) {
          const unsigned short* pb = pbase + ft * 16 * K;
          v16bf Bm = ld_frag2(pb + oB0, pb + oB1);
          acc0[ft] = __builtin_amdgcn_wmma_f32_16x16x32_bf16(
              false, A0, false, Bm, (short)0, acc0[ft], false, false);
          acc1[ft] = __builtin_amdgcn_wmma_f32_16x16x32_bf16(
              false, A1, false, Bm, (short)0, acc1[ft], false, false);
        }
      }
    }

    // ---- store C: VGPR r <-> M = r + 8*half (+16 for subtile 1), lane l15 = N ----
    float* op = out + (size_t)m0 * (G * F) + wv * F + l15;
#pragma unroll
    for (int ft = 0; ft < 8; ++ft) {
#pragma unroll
      for (int r = 0; r < 8; ++r) {
        op[(size_t)(r + (half << 3)) * (G * F) + ft * 16]        = acc0[ft][r];
        op[(size_t)(r + (half << 3) + 16) * (G * F) + ft * 16]   = acc1[ft][r];
      }
    }
  }
}

extern "C" void kernel_launch(void* const* d_in, const int* in_sizes, int n_in,
                              void* d_out, int out_size, void* d_ws, size_t ws_size,
                              hipStream_t stream) {
  (void)in_sizes; (void)n_in; (void)out_size; (void)d_ws; (void)ws_size;
  const float* fin = (const float*)d_in[0];   // (8,4096,8,128) f32
  const float* psi = (const float*)d_in[1];   // (8,128,128)    f32
  float* out = (float*)d_out;                 // (8,4096,8,128) f32
  hipLaunchKernelGGL(gconv_wmma, dim3(NBLK), dim3(256), 0, stream, fin, psi, out);
}
